// NetworkCore_50002009260148
// MI455X (gfx1250) — compile-verified, matrix operand order checked by reading, not run on hardware
//
#include <hip/hip_runtime.h>

#define T_LEN 512
#define B_SZ  64
#define H_SZ  512
#define KD    512          // reduction depth (D == H == 512)
#define NWG   32           // workgroups; each owns 16 hidden units
#define S_SL  16           // hidden units per WG
#define G3    48           // 3 * S_SL gate rows per WG (r,z,n)
#define KSTR  520          // padded bf16 row stride in LDS (16B-aligned rows, 4-bank lane step)
#define NTHR  256          // 8 waves of 32

typedef __attribute__((ext_vector_type(16))) __bf16 bf16x16;
typedef __attribute__((ext_vector_type(8)))  float  f32x8;

union FragU { uint4 u[2]; bf16x16 v; };

__device__ __forceinline__ unsigned short f2bf(float f) {
  unsigned int u = __float_as_uint(f);
  u += 0x7FFFu + ((u >> 16) & 1u);          // round-to-nearest-even
  return (unsigned short)(u >> 16);
}

__device__ __forceinline__ float sigf(float x) { return 1.0f / (1.0f + __expf(-x)); }

__global__ void cvt_bf16_kernel(const float* __restrict__ src,
                                unsigned short* __restrict__ dst, int n) {
  for (int i = blockIdx.x * blockDim.x + threadIdx.x; i < n; i += gridDim.x * blockDim.x)
    dst[i] = f2bf(src[i]);
}

// Broadcast h0 over batch into fp32 state buffer + bf16 GEMM-operand buffer; reset barrier.
__global__ void init_state_kernel(const float* __restrict__ h0l,
                                  float* __restrict__ hbuf0,
                                  unsigned short* __restrict__ hbf0,
                                  int* __restrict__ bar) {
  int i = blockIdx.x * blockDim.x + threadIdx.x;
  if (i < B_SZ * H_SZ) {
    float v = h0l[i & (H_SZ - 1)];
    hbuf0[i] = v;
    hbf0[i]  = f2bf(v);   // at t=0 the reset lerp is identity (h == h0)
  }
  if (i < 2) bar[i] = 0;
}

__global__ __launch_bounds__(NTHR, 1)
void gru_layer_kernel(const unsigned short* __restrict__ inbf,   // [T][B][512] bf16 input stream
                      const float* __restrict__ w_ih,            // [3H][H] this layer
                      const float* __restrict__ w_hh,
                      const float* __restrict__ b_ih,            // [3H]
                      const float* __restrict__ b_hh,
                      const float* __restrict__ h0l,             // [H]
                      const int*   __restrict__ fin,             // [T][B]
                      float* __restrict__ hbuf0, float* __restrict__ hbuf1,         // fp32 state ping-pong
                      unsigned short* __restrict__ hbf0, unsigned short* __restrict__ hbf1, // bf16 reset-applied ping-pong
                      unsigned short* __restrict__ outbf,        // layer0: bf16 stream for layer1 (else null)
                      float* __restrict__ outf,                  // layer1: fp32 d_out (else null)
                      float* __restrict__ hlast,                 // d_out tail slice for this layer
                      int* __restrict__ bar) {
  // Resident weight slices (bf16) + per-step gate buffers. ~122 KB of 320 KB WGP LDS.
  __shared__ __align__(16) unsigned short sh_wi[G3 * KSTR];
  __shared__ __align__(16) unsigned short sh_wh[G3 * KSTR];
  __shared__ __align__(16) float sh_gi[B_SZ * G3];
  __shared__ __align__(16) float sh_gh[B_SZ * G3];
  __shared__ float sh_bi[G3], sh_bh[G3];

  const int tid   = threadIdx.x;
  const int wg    = blockIdx.x;
  const int wave  = tid >> 5;
  const int lane  = tid & 31;
  const int m0    = (wave >> 1) * 16;   // batch-tile row base (4 m-tiles x 2 gemm halves = 8 waves)
  const int is_gh = wave & 1;           // even waves: gi GEMM; odd waves: gh GEMM
  const int lhalf = (lane & 16) ? 1 : 0;
  const int l15   = lane & 15;

  // Load this WG's weight rows (gate g, unit u = wg*16 + r) into LDS as bf16, padded rows.
  for (int idx = tid; idx < G3 * KD; idx += NTHR) {
    int r = idx >> 9;
    int k = idx & (KD - 1);
    int g = r >> 4;
    int u = wg * S_SL + (r & 15);
    size_t gro = ((size_t)(g * H_SZ + u)) * H_SZ + k;
    sh_wi[r * KSTR + k] = f2bf(w_ih[gro]);
    sh_wh[r * KSTR + k] = f2bf(w_hh[gro]);
  }
  if (tid < G3) {
    int g = tid >> 4;
    int u = wg * S_SL + (tid & 15);
    sh_bi[tid] = b_ih[g * H_SZ + u];
    sh_bh[tid] = b_hh[g * H_SZ + u];
  }
  __syncthreads();

  for (int t = 0; t < T_LEN; ++t) {
    const float*          hread = (t & 1) ? hbuf1 : hbuf0;
    float*                hwrite= (t & 1) ? hbuf0 : hbuf1;
    const unsigned short* hbfr  = (t & 1) ? hbf1 : hbf0;   // bf16(reset-applied h_{t-1})
    unsigned short*       hbfw  = (t & 1) ? hbf0 : hbf1;

    // ---- WMMA GEMMs: gates[64 x 48] += A[64 x 512] * W^T[512 x 48] ----
    f32x8 accs[3] = { (f32x8){}, (f32x8){}, (f32x8){} };
    const int bA = m0 + l15;
    const unsigned short* abase =
        is_gh ? (hbfr + bA * KD)
              : (inbf + ((size_t)t * B_SZ + bA) * KD);
    const unsigned short* wbase = is_gh ? sh_wh : sh_wi;

    #pragma unroll
    for (int ks = 0; ks < KD / 32; ++ks) {
      int kb = ks * 32 + lhalf * 8;              // per-ISA half-wave K split
      FragU fa;
      fa.u[0] = *(const uint4*)(abase + kb);
      fa.u[1] = *(const uint4*)(abase + kb + 16);
      #pragma unroll
      for (int nt = 0; nt < 3; ++nt) {
        const unsigned short* q = wbase + (nt * 16 + l15) * KSTR + kb;
        FragU fb;
        fb.u[0] = *(const uint4*)q;
        fb.u[1] = *(const uint4*)(q + 16);
        accs[nt] = __builtin_amdgcn_wmma_f32_16x16x32_bf16(
            false, fa.v, false, fb.v, (short)0, accs[nt], false, false);
      }
    }

    // C/D layout: vgpr i -> row m0+i (lanes 0-15) / m0+i+8 (lanes 16-31), col = n-tile*16 + lane%16
    float* gout = is_gh ? sh_gh : sh_gi;
    #pragma unroll
    for (int nt = 0; nt < 3; ++nt) {
      int col = nt * 16 + l15;
      #pragma unroll
      for (int i = 0; i < 8; ++i) {
        int rb = m0 + i + lhalf * 8;
        gout[rb * G3 + col] = accs[nt][i];
      }
    }
    __syncthreads();

    // ---- gates + state update for this WG's 64x16 (b, j) block ----
    for (int idx = tid; idx < B_SZ * S_SL; idx += NTHR) {
      int b = idx >> 4;
      int j = idx & 15;
      float rv = sigf(sh_gi[b * G3 + j]      + sh_bi[j]      + sh_gh[b * G3 + j]      + sh_bh[j]);
      float zv = sigf(sh_gi[b * G3 + 16 + j] + sh_bi[16 + j] + sh_gh[b * G3 + 16 + j] + sh_bh[16 + j]);
      float nv = tanhf(sh_gi[b * G3 + 32 + j] + sh_bi[32 + j] +
                       rv * (sh_gh[b * G3 + 32 + j] + sh_bh[32 + j]));
      int u = wg * S_SL + j;
      float hold = hread[b * H_SZ + u];
      float f  = (float)fin[t * B_SZ + b];
      float he = hold + f * (h0l[u] - hold);            // episode-reset lerp
      float hn = (1.0f - zv) * nv + zv * he;
      hwrite[b * H_SZ + u] = hn;
      size_t oo = ((size_t)t * B_SZ + b) * H_SZ + u;
      if (outbf) outbf[oo] = f2bf(hn);
      if (outf)  outf[oo]  = hn;
      if (t + 1 < T_LEN) {
        // pre-apply next step's reset so step t+1 reads its GEMM operand directly
        float fn  = (float)fin[(t + 1) * B_SZ + b];
        float hen = hn + fn * (h0l[u] - hn);
        hbfw[b * H_SZ + u] = f2bf(hen);
      } else {
        hlast[b * H_SZ + u] = hn;
      }
    }

    // ---- device-scope grid barrier (one per step: the serial critical path) ----
    __syncthreads();
    if (tid == 0) {
      __threadfence();
      int g = __hip_atomic_load(bar + 1, __ATOMIC_RELAXED, __HIP_MEMORY_SCOPE_AGENT);
      int a = __hip_atomic_fetch_add(bar, 1, __ATOMIC_ACQ_REL, __HIP_MEMORY_SCOPE_AGENT);
      if (a == NWG - 1) {
        __hip_atomic_store(bar, 0, __ATOMIC_RELAXED, __HIP_MEMORY_SCOPE_AGENT);
        __hip_atomic_fetch_add(bar + 1, 1, __ATOMIC_RELEASE, __HIP_MEMORY_SCOPE_AGENT);
      } else {
        while (__hip_atomic_load(bar + 1, __ATOMIC_ACQUIRE, __HIP_MEMORY_SCOPE_AGENT) == g)
          __builtin_amdgcn_s_sleep(2);
      }
    }
    __syncthreads();
  }
}

extern "C" void kernel_launch(void* const* d_in, const int* in_sizes, int n_in,
                              void* d_out, int out_size, void* d_ws, size_t ws_size,
                              hipStream_t stream) {
  (void)in_sizes; (void)n_in; (void)out_size; (void)ws_size;
  const float* x    = (const float*)d_in[0];
  const int*   fin  = (const int*)d_in[1];
  const float* w_ih = (const float*)d_in[2];
  const float* w_hh = (const float*)d_in[3];
  const float* b_ih = (const float*)d_in[4];
  const float* b_hh = (const float*)d_in[5];
  const float* h0   = (const float*)d_in[6];

  float* out   = (float*)d_out;                       // [T,B,H]
  float* hlast = out + (size_t)T_LEN * B_SZ * H_SZ;   // [L,B,H]

  char* ws = (char*)d_ws;
  size_t off = 0;
  unsigned short* xbf  = (unsigned short*)(ws + off); off += (size_t)T_LEN * B_SZ * H_SZ * 2;
  unsigned short* o0bf = (unsigned short*)(ws + off); off += (size_t)T_LEN * B_SZ * H_SZ * 2;
  float* hb0 = (float*)(ws + off); off += (size_t)B_SZ * H_SZ * 4;
  float* hb1 = (float*)(ws + off); off += (size_t)B_SZ * H_SZ * 4;
  unsigned short* hbf0 = (unsigned short*)(ws + off); off += (size_t)B_SZ * H_SZ * 2;
  unsigned short* hbf1 = (unsigned short*)(ws + off); off += (size_t)B_SZ * H_SZ * 2;
  int* bar = (int*)(ws + off); off += 256;

  // x -> bf16 stream (parallel, bandwidth-trivial)
  cvt_bf16_kernel<<<2048, NTHR, 0, stream>>>(x, xbf, T_LEN * B_SZ * H_SZ);

  // layer 0: consumes xbf, emits bf16 stream for layer 1
  init_state_kernel<<<(B_SZ * H_SZ + NTHR - 1) / NTHR, NTHR, 0, stream>>>(h0, hb0, hbf0, bar);
  gru_layer_kernel<<<NWG, NTHR, 0, stream>>>(xbf, w_ih, w_hh, b_ih, b_hh, h0, fin,
                                             hb0, hb1, hbf0, hbf1,
                                             o0bf, nullptr, hlast, bar);

  // layer 1: consumes o0bf, emits fp32 d_out
  init_state_kernel<<<(B_SZ * H_SZ + NTHR - 1) / NTHR, NTHR, 0, stream>>>(h0 + H_SZ, hb0, hbf0, bar);
  gru_layer_kernel<<<NWG, NTHR, 0, stream>>>(o0bf,
                                             w_ih + (size_t)3 * H_SZ * H_SZ,
                                             w_hh + (size_t)3 * H_SZ * H_SZ,
                                             b_ih + 3 * H_SZ, b_hh + 3 * H_SZ,
                                             h0 + H_SZ, fin,
                                             hb0, hb1, hbf0, hbf1,
                                             nullptr, out, hlast + B_SZ * H_SZ, bar);
}
